// pendulumRNNSA_20538533609627
// MI455X (gfx1250) — compile-verified, matrix operand order checked by reading, not run on hardware
//
#include <hip/hip_runtime.h>

#define Bsz  256
#define Tlen 512
#define Hd   256
#define BT   16      // batch rows per workgroup (= WMMA M)
#define ROWP 264     // padded LDS row stride in halves (breaks bank conflicts)

typedef _Float16 v16h __attribute__((ext_vector_type(16)));
typedef float    v8f  __attribute__((ext_vector_type(8)));

union Frag16 { v16h v; int4 q[2]; };

__device__ __forceinline__ v8f wmma16(v16h a, v16h b, v8f c) {
  // D = A(16x32 f16) x B(32x16 f16) + C(16x16 f32)
  return __builtin_amdgcn_wmma_f32_16x16x32_f16(false, a, false, b, (short)0, c,
                                                false, false);
}

// A fragment from LDS (row-major h state, padded rows). Lane l: m = l&15,
// halves 0..7  = k = kbase + (l>>4)*8 + 0..7
// halves 8..15 = k = kbase + 16 + (l>>4)*8 + 0..7
__device__ __forceinline__ v16h load_a_lds(const _Float16* base, int kbase, int lane) {
  const int m = lane & 15, hi = lane >> 4;
  const _Float16* p = base + m * ROWP + kbase + hi * 8;
  Frag16 f;
  f.q[0] = *(const int4*)(p);
  f.q[1] = *(const int4*)(p + 16);
  return f.v;
}

// B fragment from pre-packed fp16 weights: tile (nt,kt) stored as 32 lanes x 16
// halves contiguous (1KB blocks, fully coalesced). numKt is always 8 (K=256).
__device__ __forceinline__ v16h load_b_glb(const _Float16* __restrict__ wp,
                                           int nt, int kt, int lane) {
  const _Float16* p = wp + (((((nt << 3) + kt) << 5) + lane) << 4);
  Frag16 f;
  f.q[0] = *(const int4*)(p);
  f.q[1] = *(const int4*)(p + 8);
  return f.v;
}

__device__ __forceinline__ float fsig(float x) {
  return __fdividef(1.0f, 1.0f + __expf(-x));
}
__device__ __forceinline__ float ftanh(float x) {
  x = fminf(fmaxf(x, -12.0f), 12.0f);
  float e = __expf(2.0f * x);
  return __fdividef(e - 1.0f, e + 1.0f);
}

// ---------------- prologue: pack fp32 weights into WMMA B-fragment order ----
__global__ void pack_w_kernel(const float* __restrict__ W, _Float16* __restrict__ dst,
                              int N, int K) {
  int idx = blockIdx.x * blockDim.x + threadIdx.x;
  if (idx >= N * K) return;
  int j     = idx & 15;
  int lane  = (idx >> 4) & 31;
  int rest  = idx >> 9;
  int numKt = K >> 5;
  int kt = rest % numKt;
  int nt = rest / numKt;
  int n  = (nt << 4) + (lane & 15);
  int hi = lane >> 4;
  int krel = (j < 8) ? (hi * 8 + j) : (16 + hi * 8 + (j - 8));
  int k  = (kt << 5) + krel;
  dst[idx] = (_Float16)W[n * K + k];
}

__global__ void bias_add_kernel(const float* __restrict__ a, const float* __restrict__ b,
                                float* __restrict__ dst, int n) {
  int i = blockIdx.x * blockDim.x + threadIdx.x;
  if (i < n) dst[i] = a[i] + b[i];
}

// ---------------- main persistent recurrent kernel --------------------------
// grid = 16 workgroups (one 16-row batch tile each), 256 threads = 8 waves.
// Wave w owns h columns [w*32, w*32+32): 2 n-tiles, 4 gate blocks each.
__launch_bounds__(256)
__global__ void rnn_attn_kernel(
    const float* __restrict__ x,       // [256,512]
    const float* __restrict__ wih1,    // [1024]  (W_ih1 column)
    const float* __restrict__ bias1,   // [1024]  b_ih1+b_hh1
    const _Float16* __restrict__ wp_hh1_in,
    const _Float16* __restrict__ wp_ih2_in,
    const float* __restrict__ bias2,   // [1024]  b_ih2+b_hh2
    const _Float16* __restrict__ wp_hh2_in,
    const _Float16* __restrict__ wp_a1_in,
    const float* __restrict__ ba1,     // [256]
    const _Float16* __restrict__ wp_a2_in,
    const float* __restrict__ ba2,     // [256]
    const float* __restrict__ Wo,      // [256]
    const float* __restrict__ bo,      // [1]
    float* __restrict__ out)           // [256,512]
{
  __shared__ __align__(16) _Float16 h1s[2][BT][ROWP];
  __shared__ __align__(16) _Float16 h2s[2][BT][ROWP];
  __shared__ __align__(16) _Float16 e16[BT][ROWP];
  __shared__ float sbuf[BT][Hd];
  __shared__ float xsh[BT];
  __shared__ float pmax[BT][16];
  __shared__ float psum[BT][16];
  __shared__ float pout[BT][16];
  __shared__ float rmax[BT];

  const int tid  = threadIdx.x;
  const int wave = tid >> 5;
  const int lane = tid & 31;
  const int hi   = lane >> 4;
  const int nl   = lane & 15;
  const int wg   = blockIdx.x;

  for (int i = tid; i < BT * ROWP; i += 256) {
    (&h1s[0][0][0])[i] = (_Float16)0.0f;
    (&h2s[0][0][0])[i] = (_Float16)0.0f;
  }
  // cell state lives in registers: (m,n) <-> (lane, reg), persistent over T
  float c1r[16], c2r[16];
#pragma unroll
  for (int r = 0; r < 16; ++r) { c1r[r] = 0.0f; c2r[r] = 0.0f; }
  __syncthreads();

  // Local (mutable) copies of the packed-weight base pointers. They are
  // laundered through an empty asm each time-step so the compiler cannot
  // hoist the per-fragment global loads out of the 512-iteration loop
  // (LICM would spill ~7KB/lane of fragments to scratch). Keeping the
  // loads in-loop streams the shared weights from L2 every step.
  const _Float16* wp_hh1 = wp_hh1_in;
  const _Float16* wp_ih2 = wp_ih2_in;
  const _Float16* wp_hh2 = wp_hh2_in;
  const _Float16* wp_a1  = wp_a1_in;
  const _Float16* wp_a2  = wp_a2_in;

  int cur = 0;
  for (int t = 0; t < Tlen; ++t) {
    asm volatile("" : "+s"(wp_hh1), "+s"(wp_ih2), "+s"(wp_hh2),
                      "+s"(wp_a1), "+s"(wp_a2));

    if (tid < BT) xsh[tid] = x[(wg * BT + tid) * Tlen + t];
    __syncthreads();

    // ---------------- LSTM layer 1: gates = h1 @ W_hh1^T + x*w_ih1 + b -----
#pragma unroll
    for (int t2 = 0; t2 < 2; ++t2) {
      const int nt = wave * 2 + t2;        // n-tile within H (0..15)
      const int n  = (nt << 4) + nl;       // h column owned by this lane
      v8f ai = {}, af = {}, ag = {}, ao = {};
      const _Float16* hb = &h1s[cur][0][0];
#pragma unroll
      for (int kt = 0; kt < 8; ++kt) {
        v16h a = load_a_lds(hb, kt << 5, lane);
        ai = wmma16(a, load_b_glb(wp_hh1, nt,      kt, lane), ai);
        af = wmma16(a, load_b_glb(wp_hh1, nt + 16, kt, lane), af);
        ag = wmma16(a, load_b_glb(wp_hh1, nt + 32, kt, lane), ag);
        ao = wmma16(a, load_b_glb(wp_hh1, nt + 48, kt, lane), ao);
      }
      const float b_i = bias1[n], b_f = bias1[n + 256], b_g = bias1[n + 512], b_o = bias1[n + 768];
      const float w_i = wih1[n],  w_f = wih1[n + 256],  w_g = wih1[n + 512],  w_o = wih1[n + 768];
#pragma unroll
      for (int r = 0; r < 8; ++r) {
        const int m = r + 8 * hi;
        const float xb = xsh[m];
        const float vi = fsig (ai[r] + b_i + xb * w_i);
        const float vf = fsig (af[r] + b_f + xb * w_f);
        const float vg = ftanh(ag[r] + b_g + xb * w_g);
        const float vo = fsig (ao[r] + b_o + xb * w_o);
        const float c  = vf * c1r[t2 * 8 + r] + vi * vg;
        c1r[t2 * 8 + r] = c;
        h1s[cur ^ 1][m][n] = (_Float16)(vo * ftanh(c));
      }
    }
    __syncthreads();

    // ---------------- LSTM layer 2: gates = h1' @ W_ih2^T + h2 @ W_hh2^T ---
#pragma unroll
    for (int t2 = 0; t2 < 2; ++t2) {
      const int nt = wave * 2 + t2;
      const int n  = (nt << 4) + nl;
      v8f ai = {}, af = {}, ag = {}, ao = {};
      const _Float16* h1n = &h1s[cur ^ 1][0][0];
      const _Float16* h2c = &h2s[cur][0][0];
#pragma unroll
      for (int kt = 0; kt < 8; ++kt) {
        v16h a = load_a_lds(h1n, kt << 5, lane);
        ai = wmma16(a, load_b_glb(wp_ih2, nt,      kt, lane), ai);
        af = wmma16(a, load_b_glb(wp_ih2, nt + 16, kt, lane), af);
        ag = wmma16(a, load_b_glb(wp_ih2, nt + 32, kt, lane), ag);
        ao = wmma16(a, load_b_glb(wp_ih2, nt + 48, kt, lane), ao);
      }
#pragma unroll
      for (int kt = 0; kt < 8; ++kt) {
        v16h a = load_a_lds(h2c, kt << 5, lane);
        ai = wmma16(a, load_b_glb(wp_hh2, nt,      kt, lane), ai);
        af = wmma16(a, load_b_glb(wp_hh2, nt + 16, kt, lane), af);
        ag = wmma16(a, load_b_glb(wp_hh2, nt + 32, kt, lane), ag);
        ao = wmma16(a, load_b_glb(wp_hh2, nt + 48, kt, lane), ao);
      }
      const float b_i = bias2[n], b_f = bias2[n + 256], b_g = bias2[n + 512], b_o = bias2[n + 768];
#pragma unroll
      for (int r = 0; r < 8; ++r) {
        const int m = r + 8 * hi;
        const float vi = fsig (ai[r] + b_i);
        const float vf = fsig (af[r] + b_f);
        const float vg = ftanh(ag[r] + b_g);
        const float vo = fsig (ao[r] + b_o);
        const float c  = vf * c2r[t2 * 8 + r] + vi * vg;
        c2r[t2 * 8 + r] = c;
        h2s[cur ^ 1][m][n] = (_Float16)(vo * ftanh(c));
      }
    }
    __syncthreads();

    // ---------------- attention: e = tanh(h2' @ Wa1^T + ba1) ---------------
#pragma unroll
    for (int t2 = 0; t2 < 2; ++t2) {
      const int nt = wave * 2 + t2;
      const int n  = (nt << 4) + nl;
      v8f acc = {};
      const _Float16* h2n = &h2s[cur ^ 1][0][0];
#pragma unroll
      for (int kt = 0; kt < 8; ++kt) {
        v16h a = load_a_lds(h2n, kt << 5, lane);
        acc = wmma16(a, load_b_glb(wp_a1, nt, kt, lane), acc);
      }
      const float bb = ba1[n];
#pragma unroll
      for (int r = 0; r < 8; ++r) {
        const int m = r + 8 * hi;
        e16[m][n] = (_Float16)ftanh(acc[r] + bb);
      }
    }
    __syncthreads();

    // ---------------- attention: s = e @ Wa2^T + ba2 -----------------------
#pragma unroll
    for (int t2 = 0; t2 < 2; ++t2) {
      const int nt = wave * 2 + t2;
      const int n  = (nt << 4) + nl;
      v8f acc = {};
#pragma unroll
      for (int kt = 0; kt < 8; ++kt) {
        v16h a = load_a_lds(&e16[0][0], kt << 5, lane);
        acc = wmma16(a, load_b_glb(wp_a2, nt, kt, lane), acc);
      }
      const float bb = ba2[n];
#pragma unroll
      for (int r = 0; r < 8; ++r) {
        const int m = r + 8 * hi;
        sbuf[m][n] = acc[r] + bb;
      }
    }
    __syncthreads();

    // ---------------- softmax over H + weighted output ---------------------
    {
      const int row = tid >> 4, seg = tid & 15;
      float mx = -1e30f;
#pragma unroll
      for (int c0 = 0; c0 < 16; ++c0) mx = fmaxf(mx, sbuf[row][seg * 16 + c0]);
      pmax[row][seg] = mx;
    }
    __syncthreads();
    if (tid < BT) {
      float mx = pmax[tid][0];
#pragma unroll
      for (int s2 = 1; s2 < 16; ++s2) mx = fmaxf(mx, pmax[tid][s2]);
      rmax[tid] = mx;
    }
    __syncthreads();
    {
      const int row = tid >> 4, seg = tid & 15;
      const float mx = rmax[row];
      float sm = 0.0f, so = 0.0f;
#pragma unroll
      for (int c0 = 0; c0 < 16; ++c0) {
        const int col = seg * 16 + c0;
        const float p = __expf(sbuf[row][col] - mx);
        sm += p;
        so += p * (float)h2s[cur ^ 1][row][col] * Wo[col];
      }
      psum[row][seg] = sm;
      pout[row][seg] = so;
    }
    __syncthreads();
    if (tid < BT) {
      float sm = 0.0f, so = 0.0f;
#pragma unroll
      for (int s2 = 0; s2 < 16; ++s2) { sm += psum[tid][s2]; so += pout[tid][s2]; }
      out[(wg * BT + tid) * Tlen + t] = __fdividef(so, sm) + bo[0];
    }
    cur ^= 1;
    __syncthreads();
  }
}

extern "C" void kernel_launch(void* const* d_in, const int* in_sizes, int n_in,
                              void* d_out, int out_size, void* d_ws, size_t ws_size,
                              hipStream_t stream) {
  const float* x     = (const float*)d_in[0];
  const float* W_ih1 = (const float*)d_in[1];
  const float* b_ih1 = (const float*)d_in[2];
  const float* W_hh1 = (const float*)d_in[3];
  const float* b_hh1 = (const float*)d_in[4];
  const float* W_ih2 = (const float*)d_in[5];
  const float* b_ih2 = (const float*)d_in[6];
  const float* W_hh2 = (const float*)d_in[7];
  const float* b_hh2 = (const float*)d_in[8];
  const float* Wa1   = (const float*)d_in[9];
  const float* ba1   = (const float*)d_in[10];
  const float* Wa2   = (const float*)d_in[11];
  const float* ba2   = (const float*)d_in[12];
  const float* Wo    = (const float*)d_in[13];
  const float* bo    = (const float*)d_in[14];
  float* out = (float*)d_out;

  // workspace carve-up (fp16 packed weights + fused biases), ~1.84 MB
  _Float16* wp_hh1 = (_Float16*)d_ws;
  _Float16* wp_ih2 = wp_hh1 + 1024 * 256;
  _Float16* wp_hh2 = wp_ih2 + 1024 * 256;
  _Float16* wp_a1  = wp_hh2 + 1024 * 256;
  _Float16* wp_a2  = wp_a1 + 256 * 256;
  float*    bias1  = (float*)(wp_a2 + 256 * 256);
  float*    bias2  = bias1 + 1024;

  pack_w_kernel<<<1024, 256, 0, stream>>>(W_hh1, wp_hh1, 1024, 256);
  pack_w_kernel<<<1024, 256, 0, stream>>>(W_ih2, wp_ih2, 1024, 256);
  pack_w_kernel<<<1024, 256, 0, stream>>>(W_hh2, wp_hh2, 1024, 256);
  pack_w_kernel<<<256,  256, 0, stream>>>(Wa1,   wp_a1,  256,  256);
  pack_w_kernel<<<256,  256, 0, stream>>>(Wa2,   wp_a2,  256,  256);
  bias_add_kernel<<<4, 256, 0, stream>>>(b_ih1, b_hh1, bias1, 1024);
  bias_add_kernel<<<4, 256, 0, stream>>>(b_ih2, b_hh2, bias2, 1024);

  rnn_attn_kernel<<<16, 256, 0, stream>>>(x, W_ih1, bias1, wp_hh1, wp_ih2,
                                          bias2, wp_hh2, wp_a1, ba1, wp_a2, ba2,
                                          Wo, bo, out);
}
